// Block_21612275433892
// MI455X (gfx1250) — matrix operand order repaired and checked by reading, not compiled, not run on hardware
//
#include <hip/hip_runtime.h>
#include <cstdint>
#include <cstddef>

// ---- problem constants (match reference) ----
#define B_     4
#define C_     96
#define HEADS_ 3
#define HD_    32
#define H_     56
#define W_     56
#define HW_    (H_ * W_)          // 3136
#define NPIX_  (B_ * HW_)         // 12544
#define FFN_   384
#define QKV_STRIDE_ 288
#define PTILES_ (NPIX_ / 16)      // 784
#define PBLK_   (PTILES_ / 4)     // 196 (4 waves / block, one ptile each)

typedef __attribute__((ext_vector_type(16))) _Float16 v16h;
typedef __attribute__((ext_vector_type(8)))  float    v8f;
typedef __attribute__((ext_vector_type(4)))  float    v4f;
typedef __attribute__((ext_vector_type(4)))  unsigned int v4u;
typedef __attribute__((ext_vector_type(4)))  int      v4i;

#define AS1 __attribute__((address_space(1)))
#define AS3 __attribute__((address_space(3)))

#if __has_builtin(__builtin_amdgcn_global_load_async_to_lds_b128) && \
    __has_builtin(__builtin_amdgcn_s_wait_asynccnt)
#define USE_ASYNC_COPY 1
#else
#define USE_ASYNC_COPY 0
#endif

// ---------------------------------------------------------------------------
// weight f32 -> f16 pre-conversion (run once per call; weights reused 784x)
// ---------------------------------------------------------------------------
__global__ void cvt_f16_kernel(const float* __restrict__ src,
                               _Float16* __restrict__ dst, int n) {
  int i = blockIdx.x * blockDim.x + threadIdx.x;
  if (i < n) dst[i] = (_Float16)src[i];
}

// ---------------------------------------------------------------------------
// Kernel 1: x1 = x + depthwise3x3(x) + bias   (NCHW)
// ---------------------------------------------------------------------------
__global__ void pos_dw3_kernel(const float* __restrict__ x,
                               const float* __restrict__ w,
                               const float* __restrict__ bias,
                               float* __restrict__ out) {
  int i = blockIdx.x * blockDim.x + threadIdx.x;
  if (i >= B_ * C_ * HW_) return;
  int wx = i % W_;
  int hy = (i / W_) % H_;
  int c  = (i / HW_) % C_;
  const float* wc = w + c * 9;
  float s = bias[c];
  int plane = i - hy * W_ - wx;
#pragma unroll
  for (int dy = 0; dy < 3; ++dy) {
    int yy = hy + dy - 1;
    if ((unsigned)yy >= (unsigned)H_) continue;
#pragma unroll
    for (int dx = 0; dx < 3; ++dx) {
      int xx = wx + dx - 1;
      if ((unsigned)xx >= (unsigned)W_) continue;
      s += wc[dy * 3 + dx] * x[plane + yy * W_ + xx];
    }
  }
  out[i] = x[i] + s;
}

// ---------------------------------------------------------------------------
// Kernel 2: per-pixel channel LayerNorm (NCHW in) -> f16 pixel-major [N, C]
// ---------------------------------------------------------------------------
__global__ void ln_kernel(const float* __restrict__ x,
                          const float* __restrict__ g,
                          const float* __restrict__ bta,
                          _Float16* __restrict__ y) {
  int p = blockIdx.x * blockDim.x + threadIdx.x;
  if (p >= NPIX_) return;
  int b = p / HW_, hw = p - b * HW_;
  const float* xp = x + (size_t)b * C_ * HW_ + hw;
  float mu = 0.f;
#pragma unroll 8
  for (int c = 0; c < C_; ++c) mu += xp[(size_t)c * HW_];
  mu *= (1.0f / C_);
  float var = 0.f;
#pragma unroll 8
  for (int c = 0; c < C_; ++c) { float d = xp[(size_t)c * HW_] - mu; var += d * d; }
  var *= (1.0f / C_);
  float rs = rsqrtf(var + 1e-6f);
  _Float16* yp = y + (size_t)p * C_;
#pragma unroll 8
  for (int c = 0; c < C_; ++c)
    yp[c] = (_Float16)((xp[(size_t)c * HW_] - mu) * rs * g[c] + bta[c]);
}

// ---------------------------------------------------------------------------
// Kernel 3: WMMA GEMM, templated on K, M and epilogue MODE.
//   out[p, o] = sum_k A[p,k] * Wh[o,k] + bias[o]
//   A:  f16 pixel-major [N, K];  Wh: f16 row-major [M, K] (pre-converted)
//   MODE 0: write f32 pixel-major;  MODE 1: GELU -> f16 pm;
//   MODE 2: + NCHW residual (M == C), b128-vectorized NCHW store.
// Block = 4 waves sharing one 16-out-channel tile; B tile staged in LDS via
// gfx1250 async global->LDS (ASYNCcnt path).  K fully unrolled -> K/32 WMMAs.
// ---------------------------------------------------------------------------
template <int K, int M, int MODE>
__global__ void gemm_wmma_kernel(const _Float16* __restrict__ A,
                                 const _Float16* __restrict__ Wh,
                                 const float* __restrict__ bias,
                                 float* __restrict__ out_f32,
                                 _Float16* __restrict__ out_f16,
                                 const float* __restrict__ res_nchw,
                                 float* __restrict__ out_nchw) {
  constexpr int KP = K + 8;                 // padded LDS row (halves)
  __shared__ _Float16 bsh[16 * KP];

  const int lane  = threadIdx.x & 31;
  const int wav   = threadIdx.x >> 5;
  const int bp    = blockIdx.x % PBLK_;     // pixel-tile group
  const int ot    = blockIdx.x / PBLK_;     // out-channel tile (shared by block)
  const int p0    = (bp * 4 + wav) << 4;
  const int o0    = ot << 4;
  const int idx16 = lane & 15;              // pixel row (A) / out channel (B,D)
  const int hf    = lane >> 4;

  // ---- stage B tile (16 rows x K halves) into LDS, cooperatively ----
  {
    const _Float16* wsrc = Wh + (size_t)o0 * K;
    constexpr int VPR = K / 8;              // 16B vectors per row
    for (int idx = threadIdx.x; idx < 16 * VPR; idx += 128) {
      int r = idx / VPR, cvec = idx - r * VPR;
      const _Float16* src = wsrc + r * K + cvec * 8;
      _Float16* dst = bsh + r * KP + cvec * 8;
#if USE_ASYNC_COPY
      __builtin_amdgcn_global_load_async_to_lds_b128(
          (AS1 v4i*)(uintptr_t)src,
          (AS3 v4i*)(uint32_t)(uintptr_t)dst, 0, 0);
#else
      *(v4u*)dst = *(const v4u*)src;
#endif
    }
#if USE_ASYNC_COPY
    __builtin_amdgcn_s_wait_asynccnt(0);
#endif
    __syncthreads();
  }

  const _Float16* arow = A + (size_t)(p0 + idx16) * K;
  __builtin_prefetch(arow, 0, 1);           // global_prefetch_b8
  const _Float16* brow = bsh + idx16 * KP + hf * 16;

  v8f acc = {};
#pragma unroll
  for (int k0 = 0; k0 < K; k0 += 32) {
    v16h a, bm;
#pragma unroll
    for (int j = 0; j < 8; ++j)  a[j]     = arow[k0 + hf * 16 + j];
#pragma unroll
    for (int j = 0; j < 8; ++j)  a[8 + j] = arow[k0 + hf * 16 + 8 + j];
#pragma unroll
    for (int j = 0; j < 16; ++j) bm[j]    = brow[k0 + j];
    acc = __builtin_amdgcn_wmma_f32_16x16x32_f16(
        /*neg_a=*/false, a, /*neg_b=*/false, bm,
        /*c_mod=*/(short)0, acc, /*reuse_a=*/false, /*reuse_b=*/false);
  }

  const int o = o0 + idx16;
  const float bo = bias[o];

  if (MODE == 2) {
    // consecutive r -> consecutive hw addresses: vectorize 2 x b128
    const int bb = p0 / HW_;
    const int hwb = p0 - bb * HW_;
    size_t base = ((size_t)bb * M + o) * HW_ + hwb + hf * 8;
    v4f r0 = *(const v4f*)(res_nchw + base);
    v4f r1 = *(const v4f*)(res_nchw + base + 4);
    v4f s0, s1;
#pragma unroll
    for (int j = 0; j < 4; ++j) {
      s0[j] = acc[j] + bo + r0[j];
      s1[j] = acc[4 + j] + bo + r1[j];
    }
    *(v4f*)(out_nchw + base) = s0;
    *(v4f*)(out_nchw + base + 4) = s1;
  } else {
    const size_t base = (size_t)(p0 + hf * 8) * M + o;   // + r*M per row (imm)
#pragma unroll
    for (int r = 0; r < 8; ++r) {
      float v = acc[r] + bo;
      if (MODE == 0) {
        out_f32[base + (size_t)r * M] = v;
      } else {
        float gl = 0.5f * v * (1.0f + erff(v * 0.70710678118654752f));
        out_f16[base + (size_t)r * M] = (_Float16)gl;
      }
    }
  }
}

// ---------------------------------------------------------------------------
// Kernel 4: lepe = depthwise5x5(v part of qkv, pixel-major) -> f32 [N, C]
// ---------------------------------------------------------------------------
__global__ void lepe_dw5_kernel(const float* __restrict__ qkv,
                                const float* __restrict__ w,
                                const float* __restrict__ bias,
                                float* __restrict__ out) {
  int i = blockIdx.x * blockDim.x + threadIdx.x;   // i = p*96 + c
  if (i >= NPIX_ * C_) return;
  int c = i % C_;
  int p = i / C_;
  int wx = p % W_;
  int hy = (p / W_) % H_;
  int b  = p / HW_;
  const float* wc = w + c * 25;
  float s = bias[c];
  for (int dy = 0; dy < 5; ++dy) {
    int yy = hy + dy - 2;
    if ((unsigned)yy >= (unsigned)H_) continue;
    for (int dx = 0; dx < 5; ++dx) {
      int xx = wx + dx - 2;
      if ((unsigned)xx >= (unsigned)W_) continue;
      int pn = b * HW_ + yy * W_ + xx;
      s += wc[dy * 5 + dx] * qkv[(size_t)pn * QKV_STRIDE_ + 2 * C_ + c];
    }
  }
  out[i] = s;
}

// ---------------------------------------------------------------------------
// Kernel 5: neighborhood attention (NATTEN), online softmax, per (pixel, head)
// ---------------------------------------------------------------------------
__device__ __forceinline__ int na_start(int i, int L, int k, int d, int& g) {
  g = i % d;
  int pp = i / d;
  int Lg = (L - g + d - 1) / d;
  int s = pp - (k >> 1);
  if (s < 0) s = 0;
  if (s > Lg - k) s = Lg - k;
  return s;
}

__global__ void na_kernel(const float* __restrict__ qkv,
                          const float* __restrict__ vsrc, int vstride, int voff,
                          float* __restrict__ out,
                          int kern, int dil, float scale) {
  int t = blockIdx.x * blockDim.x + threadIdx.x;
  if (t >= B_ * HEADS_ * HW_) return;
  int wx = t % W_;
  int hy = (t / W_) % H_;
  int n  = (t / HW_) % HEADS_;
  int b  = t / (HW_ * HEADS_);
  int p  = b * HW_ + hy * W_ + wx;

  float q[HD_];
  const float* qrow = qkv + (size_t)p * QKV_STRIDE_ + n * HD_;
#pragma unroll
  for (int c = 0; c < HD_; ++c) q[c] = qrow[c] * scale;

  int gh, gw;
  int sh = na_start(hy, H_, kern, dil, gh);
  int sw = na_start(wx, W_, kern, dil, gw);

  float m = -1e30f, ssum = 0.f;
  float acc[HD_];
#pragma unroll
  for (int c = 0; c < HD_; ++c) acc[c] = 0.f;

  for (int x = 0; x < kern; ++x) {
    int ih = gh + (sh + x) * dil;
    for (int y = 0; y < kern; ++y) {
      int iw = gw + (sw + y) * dil;
      int pn = b * HW_ + ih * W_ + iw;
      const float* krow = qkv + (size_t)pn * QKV_STRIDE_ + C_ + n * HD_;
      float d = 0.f;
#pragma unroll
      for (int c = 0; c < HD_; ++c) d += q[c] * krow[c];
      float mn   = fmaxf(m, d);
      float corr = __expf(m - mn);
      float pw   = __expf(d - mn);
      ssum = ssum * corr + pw;
      const float* vrow = vsrc + (size_t)pn * vstride + voff + n * HD_;
#pragma unroll
      for (int c = 0; c < HD_; ++c) acc[c] = acc[c] * corr + pw * vrow[c];
      m = mn;
    }
  }
  float inv = 1.0f / ssum;
  float* orow = out + (size_t)p * C_ + n * HD_;
#pragma unroll
  for (int c = 0; c < HD_; ++c) orow[c] = acc[c] * inv;
}

// ---------------------------------------------------------------------------
// Kernel 6: res_f16 = f16(v2 + lepe)   ([N, C] pixel-major)
// ---------------------------------------------------------------------------
__global__ void add_res_kernel(const float* __restrict__ v2,
                               const float* __restrict__ lepe,
                               _Float16* __restrict__ res) {
  int i = blockIdx.x * blockDim.x + threadIdx.x;
  if (i >= NPIX_ * C_) return;
  res[i] = (_Float16)(v2[i] + lepe[i]);
}

// ---------------------------------------------------------------------------
// launch
// ---------------------------------------------------------------------------
extern "C" void kernel_launch(void* const* d_in, const int* in_sizes, int n_in,
                              void* d_out, int out_size, void* d_ws, size_t ws_size,
                              hipStream_t stream) {
  (void)in_sizes; (void)n_in; (void)out_size; (void)ws_size;
  const float* x      = (const float*)d_in[0];
  const float* pos_w  = (const float*)d_in[1];
  const float* pos_b  = (const float*)d_in[2];
  const float* n1_w   = (const float*)d_in[3];
  const float* n1_b   = (const float*)d_in[4];
  const float* qkv_w  = (const float*)d_in[5];
  const float* qkv_b  = (const float*)d_in[6];
  const float* lepe_w = (const float*)d_in[7];
  const float* lepe_b = (const float*)d_in[8];
  const float* op_w   = (const float*)d_in[9];
  const float* op_b   = (const float*)d_in[10];
  const float* n2_w   = (const float*)d_in[11];
  const float* n2_b   = (const float*)d_in[12];
  const float* fc1_w  = (const float*)d_in[13];
  const float* fc1_b  = (const float*)d_in[14];
  const float* fc2_w  = (const float*)d_in[15];
  const float* fc2_b  = (const float*)d_in[16];
  float* out = (float*)d_out;

  // workspace carve-up (256B aligned slices)
  char* ws = (char*)d_ws;
  size_t off = 0;
  auto carve = [&](size_t bytes) {
    char* ptr = ws + off;
    off = (off + bytes + 255) & ~(size_t)255;
    return ptr;
  };
  float*    x1     = (float*)   carve((size_t)NPIX_ * C_   * 4);
  _Float16* y16    = (_Float16*)carve((size_t)NPIX_ * C_   * 2);
  float*    qkv    = (float*)   carve((size_t)NPIX_ * 288  * 4);
  float*    lepe   = (float*)   carve((size_t)NPIX_ * C_   * 4);
  float*    v1     = (float*)   carve((size_t)NPIX_ * C_   * 4);
  float*    v2     = (float*)   carve((size_t)NPIX_ * C_   * 4);
  _Float16* res16  = (_Float16*)carve((size_t)NPIX_ * C_   * 2);
  float*    x2     = (float*)   carve((size_t)NPIX_ * C_   * 4);
  _Float16* hdn16  = (_Float16*)carve((size_t)NPIX_ * FFN_ * 2);
  _Float16* qkv_wh = (_Float16*)carve((size_t)288 * 96 * 2);
  _Float16* op_wh  = (_Float16*)carve((size_t)96  * 96 * 2);
  _Float16* fc1_wh = (_Float16*)carve((size_t)384 * 96 * 2);
  _Float16* fc2_wh = (_Float16*)carve((size_t)96 * 384 * 2);

  const float scale = 0.17677669529663687f;  // 32^-0.5

  // 0) pre-convert weights to f16 (tiny; reused 784x by the GEMMs)
  cvt_f16_kernel<<<(288 * 96 + 255) / 256, 256, 0, stream>>>(qkv_w, qkv_wh, 288 * 96);
  cvt_f16_kernel<<<(96 * 96 + 255) / 256, 256, 0, stream>>>(op_w, op_wh, 96 * 96);
  cvt_f16_kernel<<<(384 * 96 + 255) / 256, 256, 0, stream>>>(fc1_w, fc1_wh, 384 * 96);
  cvt_f16_kernel<<<(96 * 384 + 255) / 256, 256, 0, stream>>>(fc2_w, fc2_wh, 96 * 384);

  // 1) x1 = x + pos(x)
  pos_dw3_kernel<<<(B_ * C_ * HW_ + 255) / 256, 256, 0, stream>>>(x, pos_w, pos_b, x1);
  // 2) y = LN1(x1)
  ln_kernel<<<(NPIX_ + 255) / 256, 256, 0, stream>>>(x1, n1_w, n1_b, y16);
  // 3) qkv = y @ qkv_w^T + b       (WMMA, K=96 unrolled -> 3 wmma)
  gemm_wmma_kernel<96, 288, 0><<<PBLK_ * (288 / 16), 128, 0, stream>>>(
      y16, qkv_wh, qkv_b, qkv, nullptr, nullptr, nullptr);
  // 4) lepe = dw5x5(v)
  lepe_dw5_kernel<<<(NPIX_ * C_ + 255) / 256, 256, 0, stream>>>(qkv, lepe_w, lepe_b, lepe);
  // 5) NA pass 1: 7x7, dilation 1 (v from qkv)
  na_kernel<<<(B_ * HEADS_ * HW_ + 127) / 128, 128, 0, stream>>>(
      qkv, qkv, QKV_STRIDE_, 2 * C_, v1, 7, 1, scale);
  // 6) NA pass 2: 9x9, dilation 6 (v = pass-1 output)
  na_kernel<<<(B_ * HEADS_ * HW_ + 127) / 128, 128, 0, stream>>>(
      qkv, v1, C_, 0, v2, 9, 6, scale);
  // 7) res = v2 + lepe (f16)
  add_res_kernel<<<(NPIX_ * C_ + 255) / 256, 256, 0, stream>>>(v2, lepe, res16);
  // 8) x2 = x1 + (res @ op_w^T + b)   (WMMA, vectorized NCHW residual)
  gemm_wmma_kernel<96, 96, 2><<<PBLK_ * (96 / 16), 128, 0, stream>>>(
      res16, op_wh, op_b, nullptr, nullptr, x1, x2);
  // 9) y = LN2(x2)
  ln_kernel<<<(NPIX_ + 255) / 256, 256, 0, stream>>>(x2, n2_w, n2_b, y16);
  // 10) hdn = gelu(y @ fc1^T + b)     (WMMA, GELU epilogue, f16 out)
  gemm_wmma_kernel<96, 384, 1><<<PBLK_ * (384 / 16), 128, 0, stream>>>(
      y16, fc1_wh, fc1_b, nullptr, hdn16, nullptr, nullptr);
  // 11) out = x2 + (hdn @ fc2^T + b)  (WMMA K=384 -> 12 wmma, vectorized NCHW)
  gemm_wmma_kernel<384, 96, 2><<<PBLK_ * (96 / 16), 128, 0, stream>>>(
      hdn16, fc2_wh, fc2_b, nullptr, nullptr, x2, out);
}
